// QGRUCell_89034672046904
// MI455X (gfx1250) — matrix operand order repaired and checked by reading, not compiled, not run on hardware
//
#include <hip/hip_runtime.h>

// ---------------- CDNA5 WMMA types ----------------
typedef __attribute__((ext_vector_type(16))) __bf16 v16bf;
typedef __attribute__((ext_vector_type(8)))  float  v8f;

#define B_ROWS 8192
#define KDIM   512
#define HDIM   512
#define NDIM   1536

#define BM 128
#define BN 128
#define BK 32
#define KT_STEPS (KDIM / BK)
#define LDS_STRIDE 40   // 32 bf16 + 8 pad (16B-aligned fragments, spread banks)

union FragBF { v16bf v; uint4 q[2]; };

// ---------------- bf16 split helpers ----------------
__device__ __forceinline__ unsigned short bf16_rne(float x) {
    unsigned u = __float_as_uint(x);
    unsigned r = u + 0x7FFFu + ((u >> 16) & 1u);   // round to nearest even
    return (unsigned short)(r >> 16);
}
__device__ __forceinline__ float bf16_to_f32(unsigned short h) {
    return __uint_as_float(((unsigned)h) << 16);
}
__device__ __forceinline__ void cvt4(const float4 f, uint2& hi, uint2& lo) {
    unsigned short h0 = bf16_rne(f.x), h1 = bf16_rne(f.y);
    unsigned short h2 = bf16_rne(f.z), h3 = bf16_rne(f.w);
    unsigned short l0 = bf16_rne(f.x - bf16_to_f32(h0));
    unsigned short l1 = bf16_rne(f.y - bf16_to_f32(h1));
    unsigned short l2 = bf16_rne(f.z - bf16_to_f32(h2));
    unsigned short l3 = bf16_rne(f.w - bf16_to_f32(h3));
    hi = make_uint2((unsigned)h0 | ((unsigned)h1 << 16),
                    (unsigned)h2 | ((unsigned)h3 << 16));
    lo = make_uint2((unsigned)l0 | ((unsigned)l1 << 16),
                    (unsigned)l2 | ((unsigned)l3 << 16));
}

// ---------------- fixed-point helpers (match reference) ----------------
__device__ __forceinline__ float qround(float x, float s, float inv_s) {
    return floorf(x * s + 0.5f) * inv_s;
}
__device__ __forceinline__ float qsigmoid_f(float x) {
    float t = floorf(x * 134217728.0f + 0.5f);                 // Q27
    t = fminf(fmaxf(t, -2147483648.0f), 2147483647.0f);
    float s = 1.0f / (1.0f + __expf(-t * 7.450580596923828e-9f)); // /2^27
    float q31 = floorf(s * 2147483648.0f + 0.5f);              // Q31
    float q15 = floorf(q31 * 1.52587890625e-05f + 0.5f);       // requant 31->15
    return q15 * 3.0517578125e-05f;                            // /2^15
}
__device__ __forceinline__ float qtanh_f(float x) {
    float t = floorf(x * 134217728.0f + 0.5f);
    t = fminf(fmaxf(t, -2147483648.0f), 2147483647.0f);
    float v = tanhf(t * 7.450580596923828e-9f);
    float q31 = floorf(v * 2147483648.0f + 0.5f);
    float q15 = floorf(q31 * 1.52587890625e-05f + 0.5f);
    return q15 * 3.0517578125e-05f;
}

// async global->LDS copy, 16B per lane (ASYNCcnt-tracked, gfx1250)
__device__ __forceinline__ void async_copy_b128(unsigned lds_off,
                                                const unsigned short* gaddr) {
    asm volatile("global_load_async_to_lds_b128 %0, %1, off"
                 :: "v"(lds_off), "v"(gaddr) : "memory");
}
__device__ __forceinline__ void wait_async0() {
    asm volatile("s_wait_asynccnt 0x0" ::: "memory");
}
__device__ __forceinline__ void wait_ds0() {
    asm volatile("s_wait_dscnt 0x0" ::: "memory");
}
// low 32 bits of a generic pointer to __shared__ == LDS byte offset
__device__ __forceinline__ unsigned lds_off_of(const void* p) {
    return (unsigned)(uintptr_t)p;
}

// ---------------- pre-pass: f32 -> bf16 hi/lo split ----------------
__global__ __launch_bounds__(256) void split_bf16(
    const float* __restrict__ src, unsigned short* __restrict__ hi,
    unsigned short* __restrict__ lo, int n4)
{
    int i = blockIdx.x * blockDim.x + threadIdx.x;
    if (i >= n4) return;
    float4 f = ((const float4*)src)[i];
    uint2 h, l;
    cvt4(f, h, l);
    ((uint2*)hi)[i] = h;
    ((uint2*)lo)[i] = l;
}

// ---------------- GEMM: G = A @ W^T + bias, quantized to 2^-14 grid ----------
// blockIdx.z == 0 : GI = X @ Wih^T + bih ;  z == 1 : GH = H @ Whh^T + bhh
// Split-precision bf16: acc += Ahi*Whi + Ahi*Wlo + Alo*Whi (f32 accumulation).
// Tiles are DMA'd into LDS with global_load_async_to_lds_b128 and the next
// tile's DMA overlaps the current tile's WMMAs (fragments staged in VGPRs).
__global__ __launch_bounds__(256) void qgru_gemm(
    const unsigned short* __restrict__ Xhi, const unsigned short* __restrict__ Xlo,
    const unsigned short* __restrict__ Hhi, const unsigned short* __restrict__ Hlo,
    const unsigned short* __restrict__ WihHi, const unsigned short* __restrict__ WihLo,
    const unsigned short* __restrict__ WhhHi, const unsigned short* __restrict__ WhhLo,
    const float* __restrict__ bih, const float* __restrict__ bhh,
    float* __restrict__ GI, float* __restrict__ GH)
{
    __shared__ unsigned short sAhi[BM * LDS_STRIDE];
    __shared__ unsigned short sAlo[BM * LDS_STRIDE];
    __shared__ unsigned short sBhi[BN * LDS_STRIDE];
    __shared__ unsigned short sBlo[BN * LDS_STRIDE];

    const int z = blockIdx.z;
    const unsigned short* __restrict__ AhiG = z ? Hhi : Xhi;
    const unsigned short* __restrict__ AloG = z ? Hlo : Xlo;
    const unsigned short* __restrict__ BhiG = z ? WhhHi : WihHi;
    const unsigned short* __restrict__ BloG = z ? WhhLo : WihLo;
    const float* __restrict__ bs = z ? bhh : bih;
    float* __restrict__       C  = z ? GH  : GI;

    const int tid  = threadIdx.x;
    const int wave = tid >> 5;       // 0..7 : owns a 16-row strip
    const int lane = tid & 31;
    const int half = lane >> 4;
    const int mr   = lane & 15;

    const int mb = blockIdx.x * BM;
    const int nb = blockIdx.y * BN;

    // ---- async-copy plan: 2048 16B chunks (4 tiles x 128 rows x 4 segs),
    //      8 chunks per thread ----
    const unsigned short* gp[8];
    unsigned ldso[8];
#pragma unroll
    for (int i = 0; i < 8; ++i) {
        const int c    = tid + 256 * i;
        const int tile = c >> 9;          // 0:Ahi 1:Alo 2:Bhi 3:Blo
        const int w    = c & 511;
        const int row  = w >> 2;          // 0..127
        const int seg  = w & 3;           // 16B segment within the 32-col row
        const bool isB  = tile >= 2;
        const bool isLo = tile & 1;
        const unsigned short* g =
            isB ? (isLo ? BloG : BhiG) : (isLo ? AloG : AhiG);
        unsigned short* lb =
            isB ? (isLo ? sBlo : sBhi) : (isLo ? sAlo : sAhi);
        const int rb = isB ? nb : mb;
        gp[i]   = g + (size_t)(rb + row) * KDIM + seg * 8;
        ldso[i] = lds_off_of(&lb[row * LDS_STRIDE + seg * 8]);
    }

    const v8f vzero = {0.f, 0.f, 0.f, 0.f, 0.f, 0.f, 0.f, 0.f};
    v8f acc[8];
#pragma unroll
    for (int t = 0; t < 8; ++t) acc[t] = vzero;

    // A fragment (16x32 bf16): lanes 0-15 K=0..15;
    // lanes 16-31 K=16..31 (same contiguous order as B).
    const int offA0 = (wave * 16 + mr) * LDS_STRIDE + half * 16;
    const int offA1 = offA0 + 8;

    // prologue: DMA tile 0
#pragma unroll
    for (int i = 0; i < 8; ++i) async_copy_b128(ldso[i], gp[i]);

    for (int kt = 0; kt < KT_STEPS; ++kt) {
        wait_async0();          // own DMA chunks landed in LDS
        __syncthreads();        // everyone's chunks landed

        // stage all fragments of tile kt into VGPRs
        FragBF ahi, alo, bhi[8], blo[8];
        ahi.q[0] = *(const uint4*)&sAhi[offA0];
        ahi.q[1] = *(const uint4*)&sAhi[offA1];
        alo.q[0] = *(const uint4*)&sAlo[offA0];
        alo.q[1] = *(const uint4*)&sAlo[offA1];
#pragma unroll
        for (int t = 0; t < 8; ++t) {
            // B fragment (32x16): lane column = mr; lanes 0-15 K=0..15,
            // lanes 16-31 K=16..31.
            const int offB0 = (t * 16 + mr) * LDS_STRIDE + half * 16;
            bhi[t].q[0] = *(const uint4*)&sBhi[offB0];
            bhi[t].q[1] = *(const uint4*)&sBhi[offB0 + 8];
            blo[t].q[0] = *(const uint4*)&sBlo[offB0];
            blo[t].q[1] = *(const uint4*)&sBlo[offB0 + 8];
        }
        wait_ds0();             // fragments in regs, LDS reusable
        __syncthreads();

        // launch next tile's DMA; it overlaps the WMMAs below
        if (kt + 1 < KT_STEPS) {
#pragma unroll
            for (int i = 0; i < 8; ++i)
                async_copy_b128(ldso[i], gp[i] + (kt + 1) * BK);
        }

#pragma unroll
        for (int t = 0; t < 8; ++t) {
            acc[t] = __builtin_amdgcn_wmma_f32_16x16x32_bf16(
                         false, ahi.v, false, bhi[t].v, (short)0, acc[t], false, false);
            acc[t] = __builtin_amdgcn_wmma_f32_16x16x32_bf16(
                         false, ahi.v, false, blo[t].v, (short)0, acc[t], false, false);
            acc[t] = __builtin_amdgcn_wmma_f32_16x16x32_bf16(
                         false, alo.v, false, bhi[t].v, (short)0, acc[t], false, false);
        }
    }

    // Epilogue: + bias, quant_round to 2^-14 grid, store.
    // C/D layout: VGPR j -> row j (lanes 0-15) / row j+8 (lanes 16-31); lane%16 -> col.
    const int rbase = mb + wave * 16 + half * 8;
#pragma unroll
    for (int t = 0; t < 8; ++t) {
        const int col = nb + t * 16 + mr;
        const float bv = bs[col];
#pragma unroll
        for (int j = 0; j < 8; ++j) {
            float c = acc[t][j] + bv;
            c = floorf(c * 16384.0f + 0.5f) * 6.103515625e-05f;  // quant 2^14
            C[(size_t)(rbase + j) * NDIM + col] = c;
        }
    }
}

// ---------------- elementwise quantized GRU combine ----------------
__global__ __launch_bounds__(256) void qgru_elem(
    const float* __restrict__ GI, const float* __restrict__ GH,
    const float* __restrict__ Hprev, float* __restrict__ out)
{
    const int i = blockIdx.x * blockDim.x + threadIdx.x;
    const int b = i >> 9;        // /512
    const int j = i & 511;
    const size_t g = (size_t)b * NDIM;

    const float i_r = GI[g + j];
    const float i_i = GI[g + 512 + j];
    const float i_n = GI[g + 1024 + j];
    const float h_r = GH[g + j];
    const float h_i = GH[g + 512 + j];
    float       h_n = GH[g + 1024 + j];

    const float resetg = qsigmoid_f(i_r + h_r);
    const float inputg = qsigmoid_f(i_i + h_i);
    h_n = qround(h_n, 134217728.0f, 7.450580596923828e-9f);             // Q27
    const float rh = qround(resetg * h_n, 32768.0f, 3.0517578125e-05f); // Q15
    const float newg = qtanh_f(rh + i_n);
    const float nh = qround(Hprev[(size_t)b * HDIM + j], 32768.0f, 3.0517578125e-05f);
    out[(size_t)b * HDIM + j] = newg + inputg * (nh - newg);
}

extern "C" void kernel_launch(void* const* d_in, const int* in_sizes, int n_in,
                              void* d_out, int out_size, void* d_ws, size_t ws_size,
                              hipStream_t stream) {
    const float* x   = (const float*)d_in[0];
    const float* h   = (const float*)d_in[1];
    const float* wih = (const float*)d_in[2];
    const float* whh = (const float*)d_in[3];
    const float* bih = (const float*)d_in[4];
    const float* bhh = (const float*)d_in[5];
    float* out = (float*)d_out;

    const size_t nXH = (size_t)B_ROWS * KDIM;   // 4,194,304
    const size_t nW  = (size_t)NDIM * KDIM;     //   786,432
    const size_t nG  = (size_t)B_ROWS * NDIM;   // 12,582,912

    // workspace layout
    float* gi = (float*)d_ws;                  // [8192 x 1536] f32
    float* gh = gi + nG;                       // [8192 x 1536] f32
    unsigned short* p = (unsigned short*)(gh + nG);
    unsigned short* xhi = p;       p += nXH;
    unsigned short* xlo = p;       p += nXH;
    unsigned short* hhi = p;       p += nXH;
    unsigned short* hlo = p;       p += nXH;
    unsigned short* wihhi = p;     p += nW;
    unsigned short* wihlo = p;     p += nW;
    unsigned short* whhhi = p;     p += nW;
    unsigned short* whhlo = p;     // total ws ~ 140 MB

    // pre-pass: split f32 into bf16 hi/lo (bandwidth bound)
    split_bf16<<<dim3((unsigned)(nXH / 4 / 256)), dim3(256), 0, stream>>>(x, xhi, xlo, (int)(nXH / 4));
    split_bf16<<<dim3((unsigned)(nXH / 4 / 256)), dim3(256), 0, stream>>>(h, hhi, hlo, (int)(nXH / 4));
    split_bf16<<<dim3((unsigned)(nW / 4 / 256)),  dim3(256), 0, stream>>>(wih, wihhi, wihlo, (int)(nW / 4));
    split_bf16<<<dim3((unsigned)(nW / 4 / 256)),  dim3(256), 0, stream>>>(whh, whhhi, whhlo, (int)(nW / 4));

    dim3 gemm_grid(B_ROWS / BM, NDIM / BN, 2);  // (64, 12, 2)
    qgru_gemm<<<gemm_grid, dim3(256), 0, stream>>>(
        xhi, xlo, hhi, hlo, wihhi, wihlo, whhhi, whhlo, bih, bhh, gi, gh);

    const int n = B_ROWS * HDIM;                // 4.19M
    qgru_elem<<<dim3(n / 256), dim3(256), 0, stream>>>(gi, gh, h, out);
}
